// GlobalAttention_20315195310510
// MI455X (gfx1250) — matrix operand order repaired and checked by reading, not compiled, not run on hardware
//
#include <hip/hip_runtime.h>

// ---------------------------------------------------------------------------
// Types / helpers for CDNA5 (gfx1250) wave32 WMMA bf16 + TDM
// ---------------------------------------------------------------------------
typedef __attribute__((ext_vector_type(16))) __bf16 v16bf;
typedef __attribute__((ext_vector_type(8)))  __bf16 v8bf;
typedef __attribute__((ext_vector_type(8)))  float  v8f;
typedef __attribute__((ext_vector_type(4)))  unsigned int u32x4;
typedef __attribute__((ext_vector_type(8)))  int i32x8;
typedef __attribute__((ext_vector_type(4)))  int i32x4;

__device__ __forceinline__ v8f wmma_bf16(v16bf a, v16bf b, v8f c) {
    return __builtin_amdgcn_wmma_f32_16x16x32_bf16(false, a, false, b,
                                                   (short)0, c, false, false);
}
__device__ __forceinline__ v8f zero8() {
    v8f z;
#pragma unroll
    for (int i = 0; i < 8; ++i) z[i] = 0.0f;
    return z;
}
__device__ __forceinline__ v8bf load8(const __bf16* p) { return *(const v8bf*)p; }
__device__ __forceinline__ v16bf combine16(v8bf lo, v8bf hi) {
    v16bf r;
#pragma unroll
    for (int i = 0; i < 8; ++i) { r[i] = lo[i]; r[i + 8] = hi[i]; }
    return r;
}

// ---------------------------------------------------------------------------
// 16-lane max reduction on the VALU via DPP16 (no LDS unit involvement).
// xor-pattern set {1,2,7,15}: quad_perm(1,0,3,2)=0xB1, quad_perm(2,3,0,1)=0x4E,
// row_half_mirror=0x141 (i^7), row_mirror=0x140 (i^15). Each 16-lane row of
// the wave32 reduces independently = the C/D fragment row grouping.
// ---------------------------------------------------------------------------
template <int CTRL>
__device__ __forceinline__ float dpp_max_step(float x) {
    int xi = __float_as_int(x);
    int yi = __builtin_amdgcn_update_dpp(xi, xi, CTRL, 0xF, 0xF, true);
    return fmaxf(x, __int_as_float(yi));
}
__device__ __forceinline__ float rowmax16(float x) {
    x = dpp_max_step<0xB1>(x);
    x = dpp_max_step<0x4E>(x);
    x = dpp_max_step<0x141>(x);
    x = dpp_max_step<0x140>(x);
    return x;
}

// ---------------------------------------------------------------------------
// TDM: 2-D bf16 tile (tile_d0 contig elems/row, tile_d1 rows, stride0 elems)
// -> LDS at lds_off, with LDS row padding (interval 2^(c+1) DW, amount c+1 DW).
// ---------------------------------------------------------------------------
__device__ __forceinline__ void tdm_load_2d_bf16(unsigned lds_off,
                                                 const __bf16* gptr,
                                                 unsigned tile_d0, unsigned tile_d1,
                                                 unsigned stride0,
                                                 unsigned pad_int_c, unsigned pad_amt_c) {
    unsigned long long ga = (unsigned long long)(uintptr_t)gptr;
    u32x4 g0;
    g0[0] = 1u;                                   // count=1, user descriptor
    g0[1] = lds_off;                              // LDS byte address
    g0[2] = (unsigned)ga;                         // global addr lo
    g0[3] = (unsigned)((ga >> 32) & 0x01FFFFFFull) | (2u << 30);  // hi | type=2
    i32x8 g1;
    unsigned w0 = (1u << 16)                      // data_size = 2 bytes
                | (1u << 20)                      // pad_enable
                | ((pad_int_c & 7u) << 22)
                | ((pad_amt_c & 127u) << 25);
    g1[0] = (int)w0;
    g1[1] = (int)((tile_d0 & 0xFFFFu) << 16);     // tensor_dim0 lo16 @bit48
    g1[2] = (int)((tile_d1 & 0xFFFFu) << 16);     // tensor_dim0 hi=0 | tensor_dim1 lo16
    g1[3] = (int)((tile_d0 & 0xFFFFu) << 16);     // tensor_dim1 hi=0 | tile_dim0
    g1[4] = (int)(tile_d1 & 0xFFFFu);             // tile_dim1 | tile_dim2=0
    g1[5] = (int)stride0;                         // tensor_dim0_stride lo32
    g1[6] = 0;
    g1[7] = 0;
    i32x4 z4 = {0, 0, 0, 0};
#if defined(__clang_major__) && (__clang_major__ >= 23)
    i32x8 z8 = {0, 0, 0, 0, 0, 0, 0, 0};
    __builtin_amdgcn_tensor_load_to_lds(g0, g1, z4, z4, z8, 0);
#else
    __builtin_amdgcn_tensor_load_to_lds(g0, g1, z4, z4, 0);
#endif
}

// ---------------------------------------------------------------------------
// Problem constants
// ---------------------------------------------------------------------------
#define BATCH 4
#define SEQ   2048
#define DMODEL 512
#define NHEAD 8
#define DHEAD 64
#define NKV   1536

// ---------------------------------------------------------------------------
// conversions (weights stored TRANSPOSED: BT[n][k] so GEMM tiles are TDM-able)
// ---------------------------------------------------------------------------
__global__ void cvt_f32_to_bf16(const float* __restrict__ s,
                                __bf16* __restrict__ d, int n) {
    int i = (blockIdx.x * blockDim.x + threadIdx.x) * 4;
    if (i + 3 < n) {
        float4 f = *(const float4*)(s + i);
        d[i + 0] = (__bf16)f.x; d[i + 1] = (__bf16)f.y;
        d[i + 2] = (__bf16)f.z; d[i + 3] = (__bf16)f.w;
    }
}
__global__ void build_wcombT(const float* __restrict__ wq,
                             const float* __restrict__ wkv,
                             __bf16* __restrict__ d) {   // [1536][512]
    int i = blockIdx.x * blockDim.x + threadIdx.x;
    if (i >= NKV * DMODEL) return;
    int nn = i >> 9, kk = i & 511;
    float f = (nn < DMODEL) ? wq[kk * DMODEL + nn] : wkv[kk * 1024 + (nn - DMODEL)];
    d[i] = (__bf16)f;
}
__global__ void build_woT(const float* __restrict__ wo,
                          __bf16* __restrict__ d) {      // [512][512] transposed
    int i = blockIdx.x * blockDim.x + threadIdx.x;
    if (i >= DMODEL * DMODEL) return;
    int nn = i >> 9, kk = i & 511;
    d[i] = (__bf16)wo[kk * DMODEL + nn];
}

// ---------------------------------------------------------------------------
// WMMA GEMM, TDM double-buffered: C = A(Mx512) * BT^T, fp32 acc.
// Block 256 thr / 8 waves, 128x128 C tile, K-step 32, A/B tiles via TDM.
// MODE 0: scatter cols to q(B,H,N,64) (pre-scaled by 1/sqrt(d)),
//         k(B,H,N,64), v TRANSPOSED (B,H,64,N)
// MODE 1: fp32 store to out with query-row mask fused
// ---------------------------------------------------------------------------
template <int MODE>
__global__ __launch_bounds__(256)
void gemm_wmma_bf16(const __bf16* __restrict__ A, const __bf16* __restrict__ BT,
                    __bf16* __restrict__ qb, __bf16* __restrict__ kb,
                    __bf16* __restrict__ vb,
                    const unsigned char* __restrict__ mask,
                    float* __restrict__ outp) {
    const int K = DMODEL;
    const int tid  = threadIdx.x;
    const int lane = tid & 31;
    const int wid  = tid >> 5;
    const int wm   = wid >> 1;
    const int wn   = wid & 1;
    const int mbase = blockIdx.y * 128;
    const int nbase = blockIdx.x * 128;
    const int m  = lane & 15;
    const int kh = lane >> 4;

    __shared__ __bf16 As[2][128][40];      // [m][k], 80B rows (TDM-padded)
    __shared__ __bf16 Bs[2][128][40];      // [n][k]

    v8f acc[2][4];
#pragma unroll
    for (int i = 0; i < 2; ++i)
#pragma unroll
        for (int j = 0; j < 4; ++j) acc[i][j] = zero8();

    auto issue = [&](int p, int k0) {
        tdm_load_2d_bf16((unsigned)(uintptr_t)&As[p][0][0],
                         A + (size_t)mbase * K + k0, 32, 128, K, 3, 3);
        tdm_load_2d_bf16((unsigned)(uintptr_t)&Bs[p][0][0],
                         BT + (size_t)nbase * K + k0, 32, 128, K, 3, 3);
    };
    if (wid == 0) issue(0, 0);

    for (int ks = 0; ks < 16; ++ks) {
        const int p = ks & 1;
        if (wid == 0) {
            if (ks + 1 < 16) {
                issue(p ^ 1, (ks + 1) * 32);
                __builtin_amdgcn_s_wait_tensorcnt(2);  // current pair done
            } else {
                __builtin_amdgcn_s_wait_tensorcnt(0);
            }
        }
        __syncthreads();

        v16bf af[2], bfr[4];
#pragma unroll
        for (int mt = 0; mt < 2; ++mt) {
            const __bf16* ap = &As[p][wm * 32 + mt * 16 + m][0];
            af[mt] = combine16(load8(ap + kh * 16), load8(ap + kh * 16 + 8));
        }
#pragma unroll
        for (int nt = 0; nt < 4; ++nt) {
            const __bf16* bp = &Bs[p][wn * 64 + nt * 16 + m][kh * 16];
            bfr[nt] = combine16(load8(bp), load8(bp + 8));
        }
#pragma unroll
        for (int mt = 0; mt < 2; ++mt)
#pragma unroll
            for (int nt = 0; nt < 4; ++nt)
                acc[mt][nt] = wmma_bf16(af[mt], bfr[nt], acc[mt][nt]);
        __syncthreads();
    }

#pragma unroll
    for (int mt = 0; mt < 2; ++mt) {
#pragma unroll
        for (int nt = 0; nt < 4; ++nt) {
            const int C = nbase + wn * 64 + nt * 16 + m;
#pragma unroll
            for (int v = 0; v < 8; ++v) {
                const int R = mbase + wm * 32 + mt * 16 + v + 8 * kh;
                float val = acc[mt][nt][v];
                if (MODE == 0) {
                    int b = R >> 11, n = R & (SEQ - 1);
                    if (C < DMODEL) {
                        int h = C >> 6, kd = C & 63;
                        // fold attention scale 1/sqrt(64) into Q
                        qb[((((size_t)b * NHEAD + h) * SEQ + n) << 6) + kd] =
                            (__bf16)(val * 0.125f);
                    } else if (C < 2 * DMODEL) {
                        int c = C - DMODEL, h = c >> 6, kd = c & 63;
                        kb[((((size_t)b * NHEAD + h) * SEQ + n) << 6) + kd] = (__bf16)val;
                    } else {
                        int c = C - 2 * DMODEL, h = c >> 6, kd = c & 63;
                        vb[((size_t)b * NHEAD + h) * DHEAD * SEQ +
                           (size_t)kd * SEQ + n] = (__bf16)val;   // (B,H,dh,N)
                    }
                } else {
                    outp[(size_t)R * DMODEL + C] = mask[R] ? val : 0.0f;
                }
            }
        }
    }
}

// ---------------------------------------------------------------------------
// Flash attention: grid (N/128, H, B), 8 waves x 16-query tiles, 64-key steps.
// K and V tiles DMA'd by TDM (double buffered); V pre-transposed in global.
// Row sums on the matrix pipe (P @ ones); row max on the VALU via DPP16.
// ---------------------------------------------------------------------------
__global__ __launch_bounds__(256)
void attn_wmma(const __bf16* __restrict__ q, const __bf16* __restrict__ kbuf,
               const __bf16* __restrict__ vT, __bf16* __restrict__ res) {
    const int tid  = threadIdx.x;
    const int lane = tid & 31;
    const int wid  = tid >> 5;
    const int h = blockIdx.y, b = blockIdx.z;
    const int bh = b * NHEAD + h;
    const int qbase = blockIdx.x * 128 + wid * 16;
    const int m  = lane & 15;
    const int kh = lane >> 4;

    __shared__ __bf16 Ks[2][64][72];      // [key][kd], 144B rows (TDM-padded)
    __shared__ __bf16 Vt[2][64][72];      // [kd][key]
    __shared__ __bf16 Ps[8][16][72];      // per-wave P staging [qrow][key]

    const __bf16* qrow = q + ((size_t)bh * SEQ + qbase + m) * DHEAD;
    v16bf qa0 = combine16(load8(qrow + kh * 16),      load8(qrow + kh * 16 + 8));
    v16bf qa1 = combine16(load8(qrow + 32 + kh * 16), load8(qrow + 32 + kh * 16 + 8));

    v16bf ones;
#pragma unroll
    for (int i = 0; i < 16; ++i) ones[i] = (__bf16)1.0f;

    float mrun[8], lrun[8];
    v8f O[4];
#pragma unroll
    for (int i = 0; i < 8; ++i) { mrun[i] = -1e30f; lrun[i] = 0.0f; }
#pragma unroll
    for (int t = 0; t < 4; ++t) O[t] = zero8();

    auto issue = [&](int p, int s0) {
        tdm_load_2d_bf16((unsigned)(uintptr_t)&Ks[p][0][0],
                         kbuf + (size_t)bh * SEQ * DHEAD + (size_t)s0 * DHEAD,
                         64, 64, DHEAD, 4, 3);
        tdm_load_2d_bf16((unsigned)(uintptr_t)&Vt[p][0][0],
                         vT + (size_t)bh * DHEAD * SEQ + s0,
                         64, 64, SEQ, 4, 3);
    };
    if (wid == 0) issue(0, 0);

    for (int ks = 0; ks < SEQ / 64; ++ks) {
        const int p = ks & 1;
        if (wid == 0) {
            if (ks + 1 < SEQ / 64) {
                issue(p ^ 1, (ks + 1) * 64);
                __builtin_amdgcn_s_wait_tensorcnt(2);
            } else {
                __builtin_amdgcn_s_wait_tensorcnt(0);
            }
        }
        __syncthreads();

        // ---- S = Q K^T (Q pre-scaled): 4 key tiles x (2 WMMA over kd=64) ----
        v8f S[4];
#pragma unroll
        for (int sc = 0; sc < 4; ++sc) {
            const __bf16* kr = &Ks[p][sc * 16 + m][0];
            v16bf b0 = combine16(load8(kr + kh * 16),      load8(kr + kh * 16 + 8));
            v16bf b1 = combine16(load8(kr + 32 + kh * 16), load8(kr + 32 + kh * 16 + 8));
            S[sc] = wmma_bf16(qa0, b0, zero8());
            S[sc] = wmma_bf16(qa1, b1, S[sc]);
        }

        // ---- online max via DPP16 butterflies, state rescale ----
        float mnew[8], corr[8];
#pragma unroll
        for (int vv = 0; vv < 8; ++vv) {
            float mx = fmaxf(fmaxf(S[0][vv], S[1][vv]), fmaxf(S[2][vv], S[3][vv]));
            mx = rowmax16(mx);
            mnew[vv] = fmaxf(mrun[vv], mx);
            corr[vv] = __expf(mrun[vv] - mnew[vv]);
            mrun[vv] = mnew[vv];
        }

        // ---- P = exp(S - mnew) -> LDS (A-frag re-layout), O rescale ----
#pragma unroll
        for (int vv = 0; vv < 8; ++vv) {
            const int rI = vv + 8 * kh;
#pragma unroll
            for (int sc = 0; sc < 4; ++sc) {
                float pexp = __expf(S[sc][vv] - mnew[vv]);
                Ps[wid][rI][sc * 16 + m] = (__bf16)pexp;
            }
#pragma unroll
            for (int t = 0; t < 4; ++t) O[t][vv] *= corr[vv];
        }
        asm volatile("s_wait_dscnt 0" ::: "memory");   // same-wave LDS RAW

        const __bf16* pp = &Ps[wid][m][0];
        v16bf pa0 = combine16(load8(pp + kh * 16),      load8(pp + kh * 16 + 8));
        v16bf pa1 = combine16(load8(pp + 32 + kh * 16), load8(pp + 32 + kh * 16 + 8));

        // ---- row sums on the matrix pipe: RS = P @ ones ----
        v8f RS = wmma_bf16(pa0, ones, zero8());
        RS = wmma_bf16(pa1, ones, RS);
#pragma unroll
        for (int vv = 0; vv < 8; ++vv)
            lrun[vv] = lrun[vv] * corr[vv] + RS[vv];

        // ---- O += P V ----
#pragma unroll
        for (int t = 0; t < 4; ++t) {
            const __bf16* vp = &Vt[p][t * 16 + m][0];
            v16bf bv0 = combine16(load8(vp + kh * 16),      load8(vp + kh * 16 + 8));
            v16bf bv1 = combine16(load8(vp + 32 + kh * 16), load8(vp + 32 + kh * 16 + 8));
            O[t] = wmma_bf16(pa0, bv0, O[t]);
            O[t] = wmma_bf16(pa1, bv1, O[t]);
        }
        __syncthreads();
    }

    // ---- normalize, store to res (B,N,D) bf16 ----
#pragma unroll
    for (int t = 0; t < 4; ++t) {
#pragma unroll
        for (int vv = 0; vv < 8; ++vv) {
            const int r = qbase + vv + 8 * kh;
            float val = O[t][vv] / lrun[vv];
            res[((size_t)b * SEQ + r) * DMODEL + h * DHEAD + t * 16 + m] = (__bf16)val;
        }
    }
}

// ---------------------------------------------------------------------------
// Launch
// ---------------------------------------------------------------------------
extern "C" void kernel_launch(void* const* d_in, const int* in_sizes, int n_in,
                              void* d_out, int out_size, void* d_ws, size_t ws_size,
                              hipStream_t stream) {
    const float*         x    = (const float*)d_in[0];
    const unsigned char* mask = (const unsigned char*)d_in[1];  // jnp bool (1B)
    const float*         Wq   = (const float*)d_in[2];
    const float*         Wkv  = (const float*)d_in[3];
    const float*         Wo   = (const float*)d_in[4];
    float* out = (float*)d_out;

    char* ws = (char*)d_ws;
    __bf16* xbf    = (__bf16*)(ws);                   // 16,777,216 B
    __bf16* wcombT = (__bf16*)(ws + 16777216);        //  1,572,864 B [1536][512]
    __bf16* woT    = (__bf16*)(ws + 18350080);        //    524,288 B [512][512]
    __bf16* qb     = (__bf16*)(ws + 18874368);        //  8,388,608 B (B,H,N,64)
    __bf16* kb     = (__bf16*)(ws + 27262976);        //  8,388,608 B (B,H,N,64)
    __bf16* vbT    = (__bf16*)(ws + 35651584);        //  8,388,608 B (B,H,64,N)
    __bf16* resb   = (__bf16*)(ws + 44040192);        //  8,388,608 B (B,N,D)

    cvt_f32_to_bf16<<<4096, 256, 0, stream>>>(x, xbf, BATCH * SEQ * DMODEL);
    build_wcombT<<<3072, 256, 0, stream>>>(Wq, Wkv, wcombT);
    build_woT<<<1024, 256, 0, stream>>>(Wo, woT);

    gemm_wmma_bf16<0><<<dim3(NKV / 128, (BATCH * SEQ) / 128), 256, 0, stream>>>(
        xbf, wcombT, qb, kb, vbT, nullptr, nullptr);

    attn_wmma<<<dim3(SEQ / 128, NHEAD, BATCH), 256, 0, stream>>>(qb, kb, vbT, resb);

    gemm_wmma_bf16<1><<<dim3(DMODEL / 128, (BATCH * SEQ) / 128), 256, 0, stream>>>(
        resb, woT, nullptr, nullptr, nullptr, mask, out);
}
